// AveEncoder_58299886076481
// MI455X (gfx1250) — matrix operand order repaired and checked by reading, not compiled, not run on hardware
//
#include <hip/hip_runtime.h>
#include <hip/hip_bf16.h>

#define NN 32768
#define NE 524288
#define EPS 1e-5f

typedef __attribute__((ext_vector_type(16))) __bf16 v16bf;
typedef __attribute__((ext_vector_type(8)))  __bf16 v8bf;
typedef __attribute__((ext_vector_type(8)))  float  v8f;
typedef __attribute__((ext_vector_type(2)))  float  v2f;
typedef int v4i __attribute__((vector_size(16)));
typedef __attribute__((address_space(1))) v4i* gptr_v4i;
typedef __attribute__((address_space(3))) v4i* lptr_v4i;

#if defined(__has_builtin)
#if __has_builtin(__builtin_amdgcn_global_load_async_to_lds_b128) && \
    __has_builtin(__builtin_amdgcn_s_wait_asynccnt)
#define USE_ASYNC_LDS 1
#endif
#endif

// ---------------- utility kernels ----------------

__global__ void k_zero(float* __restrict__ p, size_t n) {
  size_t i = (size_t)blockIdx.x * blockDim.x + threadIdx.x;
  size_t stride = (size_t)gridDim.x * blockDim.x;
  for (; i < n; i += stride) p[i] = 0.0f;
}

__global__ void k_deg(const int* __restrict__ src, const int* __restrict__ dst,
                      float* __restrict__ ns, float* __restrict__ nd) {
  int e = blockIdx.x * blockDim.x + threadIdx.x;
  if (e < NE) {
    atomicAdd(&ns[src[e]], 1.0f);
    atomicAdd(&nd[dst[e]], 1.0f);
  }
}

__global__ void k_findeg(float* __restrict__ ns, float* __restrict__ nd) {
  int i = blockIdx.x * blockDim.x + threadIdx.x;
  if (i < NN) {
    float a = ns[i]; ns[i] = (a > 0.0f) ? rsqrtf(a) : 1.0f;
    float b = nd[i]; nd[i] = (b > 0.0f) ? rsqrtf(b) : 1.0f;
  }
}

// hs[i,f] = h[i,f] * ns[i], width 256, float4-vectorized (NN*64 threads)
__global__ void k_scale(const float* __restrict__ h, const float* __restrict__ ns,
                        float* __restrict__ hs) {
  size_t t = (size_t)blockIdx.x * blockDim.x + threadIdx.x;   // 0 .. NN*64-1
  float sc = ns[t >> 6];
  float4 v = *(const float4*)(h + (t << 2));
  v.x *= sc; v.y *= sc; v.z *= sc; v.w *= sc;
  *(float4*)(hs + (t << 2)) = v;
}

// 4 edges per block; 64 lanes/edge; float4 gather + 4 atomic f32 adds
__global__ void k_aggr(const int* __restrict__ src, const int* __restrict__ dst,
                       const float* __restrict__ hs, float* __restrict__ agg) {
  int e  = blockIdx.x * 4 + (threadIdx.x >> 6);
  int f4 = (threadIdx.x & 63) << 2;
  int s = src[e], d = dst[e];
  float4 v = *(const float4*)(hs + ((size_t)s << 8) + f4);
  float* ap = agg + ((size_t)d << 8) + f4;
  atomicAdd(ap + 0, v.x);
  atomicAdd(ap + 1, v.y);
  atomicAdd(ap + 2, v.z);
  atomicAdd(ap + 3, v.w);
}

// Abf[i,k] = bf16(agg[i,k] * nd[i]); 2 elems/thread -> packed cvt (NN*128 threads)
__global__ void k_absrc(const float* __restrict__ agg, const float* __restrict__ nd,
                        __bf16* __restrict__ out) {
  size_t t = (size_t)blockIdx.x * blockDim.x + threadIdx.x;   // 0 .. NN*128-1
  size_t idx = t << 1;
  float sc = nd[idx >> 8];
  out[idx]     = (__bf16)(agg[idx]     * sc);
  out[idx + 1] = (__bf16)(agg[idx + 1] * sc);
}

// Wt[n,k] = bf16(W[k,n]); K fixed 256, idx over N*256
__global__ void k_convWt(const float* __restrict__ W, __bf16* __restrict__ Wt, int N) {
  int idx = blockIdx.x * blockDim.x + threadIdx.x;
  int n = idx >> 8, k = idx & 255;
  Wt[idx] = (__bf16)(W[(size_t)k * N + n]);
}

// f32 transpose 1024x1024 for head weights: out[n,k] = in[k,n]
__global__ void k_transF(const float* __restrict__ in, float* __restrict__ out) {
  int idx = blockIdx.x * blockDim.x + threadIdx.x;
  int n = idx >> 10, k = idx & 1023;
  out[idx] = in[(size_t)k * 1024 + n];
}

// ---------------- bf16 WMMA GEMM with async-LDS A staging ----------------
// C[M,N] = Abf[M,256] @ Wt[N,256]^T ; WG = 16 rows x 128 cols, 8 waves (16x16 tile each)
#define LDS_ROW 528   // 512B row + 16B pad (bank-conflict break)

__global__ void k_gemm_bf16(const __bf16* __restrict__ A, const __bf16* __restrict__ Wt,
                            float* __restrict__ C, int N) {
  constexpr int K = 256;
  __shared__ __align__(16) char lds[16 * LDS_ROW];
  int wgPerRow = N >> 7;
  int bi = blockIdx.x / wgPerRow;
  int bj = blockIdx.x - bi * wgPerRow;
  int i0 = bi << 4;
  int jbase = bj << 7;

  // stage 16x256 bf16 A strip (8KB contiguous in global) into LDS
  {
    const char* g = (const char*)(A + (size_t)i0 * K);
    int t = threadIdx.x;
#pragma unroll
    for (int c = 0; c < 2; ++c) {
      int off = (c * 256 + t) << 4;          // 16B chunk
      int row = off >> 9;
      int laddr = row * LDS_ROW + (off & 511);
#ifdef USE_ASYNC_LDS
      __builtin_amdgcn_global_load_async_to_lds_b128(
          (gptr_v4i)(g + off),
          (lptr_v4i)(lds + laddr),
          0, 0);
#else
      float4 tmp = *(const float4*)(g + off);
      *(float4*)(lds + laddr) = tmp;
#endif
    }
#ifdef USE_ASYNC_LDS
    __builtin_amdgcn_s_wait_asynccnt(0);
#endif
    __syncthreads();
  }

  int lane = threadIdx.x & 31;
  int wv   = threadIdx.x >> 5;
  int l15  = lane & 15;
  int hi   = (lane >> 4) & 1;
  int n = jbase + (wv << 4) + l15;
  const __bf16* Brow = Wt + (size_t)n * K;
  v8f acc = {};
#pragma unroll
  for (int k0 = 0; k0 < K; k0 += 32) {
    int ka = k0 + hi * 8;                      // A: lanes16-31 -> K+8..15 / K+24..31
    const char* ap = lds + l15 * LDS_ROW + ka * 2;
    v8bf a0 = *(const v8bf*)ap;                // K ka..ka+7
    v8bf a1 = *(const v8bf*)(ap + 32);         // K ka+16..ka+23
    v16bf a = __builtin_shufflevector(a0, a1, 0,1,2,3,4,5,6,7,8,9,10,11,12,13,14,15);
    int kb = k0 + hi * 8;                      // B: lanes16-31 -> K+8..15 / K+24..31
    v8bf b0 = *(const v8bf*)(Brow + kb);
    v8bf b1 = *(const v8bf*)(Brow + kb + 16);
    v16bf b = __builtin_shufflevector(b0, b1, 0,1,2,3,4,5,6,7,8,9,10,11,12,13,14,15);
    acc = __builtin_amdgcn_wmma_f32_16x16x32_bf16(false, a, false, b,
                                                  (short)0, acc, false, false);
  }
  int mb = i0 + hi * 8;
  float* Crow = C + (size_t)mb * N + n;
#pragma unroll
  for (int r = 0; r < 8; ++r) Crow[(size_t)r * N] = acc[r];
}

// ---------------- f32 WMMA head GEMM: C = A[M,K] @ WtF[N,K]^T + bias ----------------
__global__ void k_gemm_f32(const float* __restrict__ A, const float* __restrict__ WtF,
                           const float* __restrict__ bias, float* __restrict__ C,
                           int K, int N) {
  int lane = threadIdx.x & 31;
  int wave = blockIdx.x * (blockDim.x >> 5) + (threadIdx.x >> 5);
  int tilesN = N >> 4;
  int ti = wave / tilesN;
  int tj = wave - ti * tilesN;
  int i0 = ti << 4, j0 = tj << 4;
  int l15 = lane & 15;
  int hi  = (lane >> 4) & 1;
  int m = i0 + l15;
  int n = j0 + l15;
  const float* Arow = A   + (size_t)m * K;
  const float* Brow = WtF + (size_t)n * K;
  v8f acc = {};
  for (int k0 = 0; k0 < K; k0 += 4) {
    int ka = k0 + hi * 2;                      // lanes0-15: K,K+1 ; lanes16-31: K+2,K+3
    v2f a = *(const v2f*)(Arow + ka);
    v2f b = *(const v2f*)(Brow + ka);
    acc = __builtin_amdgcn_wmma_f32_16x16x4_f32(false, a, false, b,
                                                (short)0, acc, false, false);
  }
  float bv = bias[n];
  int mb = i0 + hi * 8;
  float* Crow = C + (size_t)mb * N + n;
#pragma unroll
  for (int r = 0; r < 8; ++r) Crow[(size_t)r * N] = acc[r] + bv;
}

// ---------------- bias + leaky_relu + layernorm (in place, block per row) ----------------
__global__ void k_postln(float* __restrict__ h, const float* __restrict__ bias,
                         int dim, int doAct) {
  __shared__ float redS[256];
  __shared__ float redQ[256];
  int row = blockIdx.x;
  float* p = h + (size_t)row * dim;
  int per = dim >> 8;
  float vals[4];
  float s = 0.0f, sq = 0.0f;
  for (int t = 0; t < per; ++t) {
    int f = (t << 8) + threadIdx.x;
    float v = p[f];
    if (bias) v += bias[f];
    if (doAct) v = (v >= 0.0f) ? v : 0.01f * v;
    vals[t] = v;
    s += v; sq += v * v;
  }
  redS[threadIdx.x] = s; redQ[threadIdx.x] = sq;
  __syncthreads();
  for (int off = 128; off > 0; off >>= 1) {
    if (threadIdx.x < off) {
      redS[threadIdx.x] += redS[threadIdx.x + off];
      redQ[threadIdx.x] += redQ[threadIdx.x + off];
    }
    __syncthreads();
  }
  float inv  = 1.0f / (float)dim;
  float mu   = redS[0] * inv;
  float var  = redQ[0] * inv - mu * mu;
  float rstd = rsqrtf(var + EPS);
  for (int t = 0; t < per; ++t) {
    int f = (t << 8) + threadIdx.x;
    p[f] = (vals[t] - mu) * rstd;
  }
}

// mean over 256 nodes per graph: h[32768,1024] -> out[128,1024]
__global__ void k_pool(const float* __restrict__ h, float* __restrict__ out) {
  int b = blockIdx.x;
  for (int t = 0; t < 4; ++t) {
    int d = (t << 8) + threadIdx.x;
    const float* p = h + ((size_t)b * 256) * 1024 + d;
    float s = 0.0f;
    for (int i = 0; i < 256; ++i) s += p[(size_t)i * 1024];
    out[(size_t)b * 1024 + d] = s * (1.0f / 256.0f);
  }
}

// ---------------- host side ----------------

extern "C" void kernel_launch(void* const* d_in, const int* in_sizes, int n_in,
                              void* d_out, int out_size, void* d_ws, size_t ws_size,
                              hipStream_t stream) {
  (void)in_sizes; (void)n_in; (void)out_size; (void)ws_size;
  const float* x   = (const float*)d_in[0];
  const int*   src = (const int*)d_in[1];
  const int*   dst = (const int*)d_in[2];
  const float* W1  = (const float*)d_in[3];  const float* b1 = (const float*)d_in[4];
  const float* W2  = (const float*)d_in[5];  const float* b2 = (const float*)d_in[6];
  const float* W3  = (const float*)d_in[7];  const float* b3 = (const float*)d_in[8];
  const float* W4  = (const float*)d_in[9];  const float* b4 = (const float*)d_in[10];
  const float* Wm  = (const float*)d_in[11]; const float* bm = (const float*)d_in[12];
  const float* Wsd = (const float*)d_in[13]; const float* bs = (const float*)d_in[14];

  float* ws     = (float*)d_ws;
  float* ns     = ws;                           // 32768
  float* nd     = ns + NN;                      // 32768
  float* hs     = nd + NN;                      // NN*256
  float* agg    = hs   + (size_t)NN * 256;      // NN*256
  float* h1     = agg  + (size_t)NN * 256;      // NN*256
  float* h2     = h1   + (size_t)NN * 256;      // NN*256
  float* hbig   = h2   + (size_t)NN * 256;      // NN*1024
  float* pooled = hbig + (size_t)NN * 1024;     // 128*1024
  float* WmT    = pooled + 128 * 1024;          // 1024*1024
  float* WsT    = WmT + 1024 * 1024;            // 1024*1024
  __bf16* Abf   = (__bf16*)(WsT + 1024 * 1024); // NN*256 bf16
  __bf16* WtB   = Abf + (size_t)NN * 256;       // up to 1024*256 bf16
  float* outp = (float*)d_out;

  // degree normalizers
  k_zero<<<256, 256, 0, stream>>>(ns, (size_t)2 * NN);
  k_deg<<<NE / 256, 256, 0, stream>>>(src, dst, ns, nd);
  k_findeg<<<NN / 256, 256, 0, stream>>>(ns, nd);

  // head weight transposes (independent of layer pipeline)
  k_transF<<<4096, 256, 0, stream>>>(Wm,  WmT);
  k_transF<<<4096, 256, 0, stream>>>(Wsd, WsT);

  struct Layer { const float* hin; const float* W; const float* bias; float* gout; int dimout; };
  Layer L[4] = {
    { x,  W1, b1, h1,   256  },
    { h1, W2, b2, h2,   256  },
    { h2, W3, b3, h1,   256  },
    { h1, W4, b4, hbig, 1024 },
  };
  for (int l = 0; l < 4; ++l) {
    int dn = L[l].dimout;
    k_scale<<<NN * 64 / 256, 256, 0, stream>>>(L[l].hin, ns, hs);
    k_zero<<<4096, 256, 0, stream>>>(agg, (size_t)NN * 256);
    k_aggr<<<NE / 4, 256, 0, stream>>>(src, dst, hs, agg);
    k_absrc<<<NN * 128 / 256, 256, 0, stream>>>(agg, nd, Abf);
    k_convWt<<<dn, 256, 0, stream>>>(L[l].W, WtB, dn);           // dn*256 elems
    int grid = (NN / 16) * (dn / 128);
    k_gemm_bf16<<<grid, 256, 0, stream>>>(Abf, WtB, L[l].gout, dn);
    k_postln<<<NN, 256, 0, stream>>>(L[l].gout, L[l].bias, dn, 1);
  }

  k_pool<<<128, 256, 0, stream>>>(hbig, pooled);
  k_postln<<<128, 256, 0, stream>>>(pooled, nullptr, 1024, 0);

  int headTiles = (128 / 16) * (1024 / 16);  // 512
  k_gemm_f32<<<headTiles / 8, 256, 0, stream>>>(pooled, WmT, bm, outp,              1024, 1024);
  k_gemm_f32<<<headTiles / 8, 256, 0, stream>>>(pooled, WsT, bs, outp + 128 * 1024, 1024, 1024);
}
